// FusionMamba_24412594111024
// MI455X (gfx1250) — matrix operand order repaired and checked by reading, not compiled, not run on hardware
//
#include <hip/hip_runtime.h>
#include <math.h>

// ============================================================================
// FusionMamba on gfx1250 (MI455X, wave32, WMMA).
// All dense projections use v_wmma_f32_16x16x32_f16 (f16 inputs, f32 accum),
// K-templated so the k-loop fully unrolls (2 or 4 static WMMAs per kernel).
// x_proj weights are pre-padded 36 -> 48 cols so every GEMM has N % 16 == 0
// and the B-fragment loads are unconditional + lane-coalesced.
// Selective scan: 16 lanes (= D_STATE) per (batch, channel); cross-lane
// reduction via shfl_xor; global_prefetch_b8 ahead of the serial walk.
// Transcendentals use hardware-native v_exp_f32/v_log_f32/v_rcp_f32 paths so
// the serial scan recurrence and activation epilogues stay short & branchless.
// Workspace requirement: ~53 MB of d_ws.
// ============================================================================

typedef __attribute__((ext_vector_type(16))) _Float16 v16h;
typedef __attribute__((ext_vector_type(8)))  float    v8f;

namespace fm {
constexpr int B     = 2;
constexpr int D     = 64;
constexpr int Himg  = 64;
constexpr int Wimg  = 64;
constexpr int L     = Himg * Wimg;   // 4096
constexpr int NTOK  = B * L;         // 8192
constexpr int NELT  = NTOK * D;      // 524288
constexpr int DSTATE = 16;
constexpr int DTRANK = 4;
constexpr int DBL_LD = 48;           // x_proj output padded 36 -> 48
}
using namespace fm;

// Fast, branchless device math (native v_exp_f32 / v_log_f32 / v_rcp_f32).
__device__ __forceinline__ float frcp(float x)  { return __builtin_amdgcn_rcpf(x); }
__device__ __forceinline__ float fexp(float x)  { return __expf(x); }
__device__ __forceinline__ float fsig(float x)  { return frcp(1.f + __expf(-x)); }
__device__ __forceinline__ float fsilu(float x) { return x * frcp(1.f + __expf(-x)); }
__device__ __forceinline__ float ftanh(float x) {
  // exact at +/-inf via v_exp_f32 saturation; no exec-mask slow path
  return 1.f - 2.f * frcp(1.f + __expf(2.f * x));
}
__device__ __forceinline__ float fsoftplus(float x) {
  return fmaxf(x, 0.f) + __logf(1.f + __expf(-fabsf(x)));
}

// ---------------------------------------------------------------------------
// Elementwise / layout kernels
// ---------------------------------------------------------------------------
__global__ void k_to_seq(const float* __restrict__ in, float* __restrict__ out, int n) {
  int i = blockIdx.x * blockDim.x + threadIdx.x;
  if (i >= n) return;
  int t = i % L; int c = (i / L) % D; int b = i / (L * D);
  out[(b * L + t) * D + c] = in[i];   // (b,c,t) -> (b,t,c)
}

__global__ void k_from_seq(const float* __restrict__ in, float* __restrict__ out, int n) {
  int i = blockIdx.x * blockDim.x + threadIdx.x;
  if (i >= n) return;
  int t = i % L; int c = (i / L) % D; int b = i / (L * D);
  out[i] = in[(b * L + t) * D + c];   // (b,t,c) -> (b,c,t)
}

// LayerNorm over 64 channels; one wave per token, 2 channels per lane.
// out = LN(x)*w + b (+ add if non-null)
__global__ void k_layernorm(const float* __restrict__ x, const float* __restrict__ w,
                            const float* __restrict__ bb, const float* __restrict__ add,
                            float* __restrict__ out, int ntok) {
  int tok  = blockIdx.x * (blockDim.x >> 5) + (threadIdx.x >> 5);
  if (tok >= ntok) return;
  int lane = threadIdx.x & 31;
  const float* xr = x + tok * D;
  float v0 = xr[lane], v1 = xr[lane + 32];
  float s = v0 + v1, sq = v0 * v0 + v1 * v1;
#pragma unroll
  for (int off = 16; off; off >>= 1) {
    s  += __shfl_xor(s,  off, 32);
    sq += __shfl_xor(sq, off, 32);
  }
  float mean = s * (1.f / 64.f);
  float var  = sq * (1.f / 64.f) - mean * mean;
  float inv  = rsqrtf(var + 1e-5f);
  float o0 = (v0 - mean) * inv * w[lane]      + bb[lane];
  float o1 = (v1 - mean) * inv * w[lane + 32] + bb[lane + 32];
  if (add) {
    o0 += add[tok * D + lane];
    o1 += add[tok * D + lane + 32];
  }
  out[tok * D + lane]      = o0;
  out[tok * D + lane + 32] = o1;
}

__global__ void k_concat(const float* __restrict__ a, const float* __restrict__ b,
                         float* __restrict__ cat, int n) {
  int i = blockIdx.x * blockDim.x + threadIdx.x;
  if (i >= n) return;
  int t = i / D, c = i % D;
  cat[t * 128 + c]      = a[i];
  cat[t * 128 + 64 + c] = b[i];
}

__global__ void k_flip(const float* __restrict__ in, float* __restrict__ out, int n) {
  int i = blockIdx.x * blockDim.x + threadIdx.x;
  if (i >= n) return;
  int d = i % D; int s = (i / D) % L; int b = i / (D * L);
  out[i] = in[(b * L + (L - 1 - s)) * D + d];
}

// Zero-pad x_proj weights [64,36] -> [64,48] so the GEMM has N % 16 == 0.
__global__ void k_pad_xproj(const float* __restrict__ w36, float* __restrict__ w48, int n) {
  int i = blockIdx.x * blockDim.x + threadIdx.x;
  if (i >= n) return;
  int k = i / DBL_LD, c = i % DBL_LD;
  w48[i] = (c < DTRANK + 2 * DSTATE) ? w36[k * (DTRANK + 2 * DSTATE) + c] : 0.f;
}

// ---------------------------------------------------------------------------
// WMMA GEMM:  C[M=8192, N] = act( A[M,K](f32,lda) @ W[K,N](f32,ldw) + bias )
// One 16x16 output tile per wave. Requirements: N % 16 == 0, K in {64,128}.
// K is a template parameter so the k-loop fully unrolls (2 or 4 WMMAs).
// All index math is 32-bit (max index 2^20). act: 0=none, 1=sigmoid, 2=gelu.
// ---------------------------------------------------------------------------
template <int K>
__global__ void k_gemm_wmma(const float* __restrict__ A, int lda,
                            const float* __restrict__ Wt, int ldw,
                            const float* __restrict__ bias,
                            float* __restrict__ C, int ldc,
                            int tiles_n, int total_tiles, int act) {
  int wave = blockIdx.x * (blockDim.x >> 5) + (threadIdx.x >> 5);
  if (wave >= total_tiles) return;               // wave-uniform branch (EXEC stays full)
  int mt = wave / tiles_n, nt = wave % tiles_n;
  int lane = threadIdx.x & 31;
  int hf   = lane >> 4;                          // half-wave id
  int lr   = lane & 15;
  int row  = (mt << 4) + lr;                     // A row for this lane
  int col  = (nt << 4) + lr;                     // B/C column for this lane
  const float* Arow = A + row * lda;

  v8f acc = {};
#pragma unroll
  for (int k0 = 0; k0 < K; k0 += 32) {
    v16h a, b;
    // A 16x32 f16 fragment: elems 0-7 -> k = k0+hf*8+e ; 8-15 -> k0+16+hf*8+e
    const float4* Ap = (const float4*)(Arow + k0 + hf * 8);
    float4 a0 = Ap[0], a1 = Ap[1];
    float4 a2 = Ap[4], a3 = Ap[5];
    a[0]  = (_Float16)a0.x; a[1]  = (_Float16)a0.y; a[2]  = (_Float16)a0.z; a[3]  = (_Float16)a0.w;
    a[4]  = (_Float16)a1.x; a[5]  = (_Float16)a1.y; a[6]  = (_Float16)a1.z; a[7]  = (_Float16)a1.w;
    a[8]  = (_Float16)a2.x; a[9]  = (_Float16)a2.y; a[10] = (_Float16)a2.z; a[11] = (_Float16)a2.w;
    a[12] = (_Float16)a3.x; a[13] = (_Float16)a3.y; a[14] = (_Float16)a3.z; a[15] = (_Float16)a3.w;
    // B 32x16 f16 fragment: elems 0-7 -> k = k0+hf*8+e ; 8-15 -> k0+16+hf*8+e, column = col.
    // At fixed e the 16 lanes of a half-wave read consecutive addresses.
    const float* Wp = Wt + (k0 + hf * 8) * ldw + col;
#pragma unroll
    for (int e = 0; e < 16; ++e)
      b[e] = (_Float16)Wp[((e & 7) + ((e >> 3) << 4)) * ldw];
    acc = __builtin_amdgcn_wmma_f32_16x16x32_f16(false, a, false, b,
                                                 (short)0, acc, false, false);
  }
  float bv = bias ? bias[col] : 0.f;
#pragma unroll
  for (int v = 0; v < 8; ++v) {
    int m = (mt << 4) + (hf << 3) + v;           // D layout: M = hf*8 + v
    float x = acc[v] + bv;
    if (act == 1) {
      x = fsig(x);
    } else if (act == 2) {
      float x3 = x * x * x;
      x = 0.5f * x * (1.f + ftanh(0.7978845608028654f * (x + 0.044715f * x3)));
    }
    C[m * ldc + col] = x;
  }
}

// ---------------------------------------------------------------------------
// Depthwise causal conv (K=4) + SiLU, with optional sequence flip folded in.
// ---------------------------------------------------------------------------
__global__ void k_dwconv_silu(const float* __restrict__ x, int ldx,
                              const float* __restrict__ Wc, const float* __restrict__ bc,
                              float* __restrict__ out, int flip, int n) {
  int i = blockIdx.x * blockDim.x + threadIdx.x;
  if (i >= n) return;
  int d = i % D; int s = (i / D) % L; int b = i / (D * L);
  float acc = bc[d];
#pragma unroll
  for (int k = 0; k < 4; ++k) {
    int j = s - 3 + k;                            // position in (possibly flipped) seq
    if (j >= 0) {
      int l = flip ? (L - 1 - j) : j;             // map back to storage order
      acc += x[(b * L + l) * ldx + d] * Wc[d * 4 + k];
    }
  }
  out[i] = fsilu(acc);
}

// dt = softplus(dbl[:, :4] @ dt_proj_W + dt_proj_b)
__global__ void k_dt(const float* __restrict__ dbl, const float* __restrict__ Wdt,
                     const float* __restrict__ bdt, float* __restrict__ dt, int n) {
  int i = blockIdx.x * blockDim.x + threadIdx.x;
  if (i >= n) return;
  int d = i % D; int t = i / D;
  float s = bdt[d];
#pragma unroll
  for (int r = 0; r < DTRANK; ++r)
    s += dbl[t * DBL_LD + r] * Wdt[r * D + d];
  dt[i] = fsoftplus(s);
}

// ---------------------------------------------------------------------------
// Selective scan. 16 lanes = 16 states for one (b, d). Sequential over L.
// ---------------------------------------------------------------------------
__global__ void k_scan(const float* __restrict__ dt, const float* __restrict__ xc,
                       const float* __restrict__ dbl, const float* __restrict__ A_log,
                       const float* __restrict__ Dp, float* __restrict__ y,
                       int flip, int accum) {
  int gid = blockIdx.x * (blockDim.x >> 4) + (threadIdx.x >> 4);
  if (gid >= B * D) return;
  int n = threadIdx.x & 15;
  int d = gid % D; int b = gid / D;
  float Aval = -fexp(A_log[d * DSTATE + n]);
  float Dv   = Dp[d];
  float h = 0.f;
  const float* dtp = dt  + b * L * D + d;
  const float* xcp = xc  + b * L * D + d;
  const float* Bp  = dbl + b * L * DBL_LD + DTRANK + n;
  const float* Cp  = dbl + b * L * DBL_LD + DTRANK + DSTATE + n;
  for (int s = 0; s < L; ++s) {
    if ((s & 31) == 0 && s + 32 < L) {            // stride-ahead prefetch hints
      __builtin_prefetch(dtp + (s + 32) * D, 0, 0);
      __builtin_prefetch(xcp + (s + 32) * D, 0, 0);
      __builtin_prefetch(Bp  + (s + 32) * DBL_LD, 0, 0);
    }
    float dtv = dtp[s * D];
    float xcv = xcp[s * D];
    float Bv  = Bp[s * DBL_LD];
    float Cv  = Cp[s * DBL_LD];
    h = fexp(dtv * Aval) * h + dtv * xcv * Bv;    // short serial chain: mul, v_exp, fma
    float p = h * Cv;
    p += __shfl_xor(p, 1, 32);
    p += __shfl_xor(p, 2, 32);
    p += __shfl_xor(p, 4, 32);
    p += __shfl_xor(p, 8, 32);
    if (n == 0) {
      int l = flip ? (L - 1 - s) : s;
      int oi = (b * L + l) * D + d;
      float val = 0.5f * (p + xcv * Dv);
      if (accum) y[oi] += val; else y[oi] = val;
    }
  }
}

// g = y * silu(z); out = rmsnorm(g) * w   (z has row stride ldz)
__global__ void k_gate_rms(const float* __restrict__ y, const float* __restrict__ z, int ldz,
                           const float* __restrict__ w, float* __restrict__ out, int ntok) {
  int tok  = blockIdx.x * (blockDim.x >> 5) + (threadIdx.x >> 5);
  if (tok >= ntok) return;
  int lane = threadIdx.x & 31;
  float y0 = y[tok * D + lane],    y1 = y[tok * D + lane + 32];
  float z0 = z[tok * ldz + lane],  z1 = z[tok * ldz + lane + 32];
  float g0 = y0 * fsilu(z0);
  float g1 = y1 * fsilu(z1);
  float sq = g0 * g0 + g1 * g1;
#pragma unroll
  for (int off = 16; off; off >>= 1) sq += __shfl_xor(sq, off, 32);
  float inv = rsqrtf(sq * (1.f / 64.f) + 1e-5f);
  out[tok * D + lane]      = g0 * inv * w[lane];
  out[tok * D + lane + 32] = g1 * inv * w[lane + 32];
}

__global__ void k_combine_cross(const float* __restrict__ t2, const float* __restrict__ wgt,
                                const float* __restrict__ n0, const float* __restrict__ x0,
                                float* __restrict__ out, int n) {
  int i = blockIdx.x * blockDim.x + threadIdx.x;
  if (i >= n) return;
  float wv = wgt[i];
  out[i] = t2[i] * wv + n0[i] * (1.f - wv) + x0[i];
}

__global__ void k_fuse(const float* __restrict__ a_, const float* __restrict__ b_,
                       const float* __restrict__ alpha, const float* __restrict__ beta,
                       float* __restrict__ out, int n) {
  int i = blockIdx.x * blockDim.x + threadIdx.x;
  if (i >= n) return;
  float A = alpha[0], Bc = beta[0];
  out[i] = (A * a_[i] + Bc * b_[i]) * frcp(A + Bc);
}

// ---------------------------------------------------------------------------
// Host side: parameter pytree parsing (JAX tree_flatten order: dict keys
// sorted lexicographically, lists in order) + orchestration.
// ---------------------------------------------------------------------------
namespace {

struct SubP   { const float *A_log, *Dp, *conv_W, *conv_b, *dt_proj_W, *dt_proj_b, *x_proj_W; };
struct MambaP { SubP bwd, fwd; const float *in_proj2_W, *in_proj_W, *norm_w, *out_proj_W; };
struct CrossP { const float *cw_W, *cw_b, *ln0_b, *ln0_w, *ln1_b, *ln1_w;
                MambaP m; const float *post_b, *post_w; };
struct SingleP{ const float *ln_b, *ln_w; MambaP m; const float *post_b, *post_w; };
struct OutP   { const float *W1, *W2, *b1, *b2, *ln_b, *ln_w; };

struct Cursor {
  void* const* din; int i;
  const float* next() { return (const float*)din[i++]; }
};

SubP read_sub(Cursor& c) {
  SubP s;                      // sorted: A_log, D, conv_W, conv_b, dt_proj_W, dt_proj_b, x_proj_W
  s.A_log = c.next(); s.Dp = c.next(); s.conv_W = c.next(); s.conv_b = c.next();
  s.dt_proj_W = c.next(); s.dt_proj_b = c.next(); s.x_proj_W = c.next();
  return s;
}
MambaP read_mamba(Cursor& c, bool cross) {
  MambaP m;                    // sorted: bwd, fwd, [in_proj2_W,] in_proj_W, norm_w, out_proj_W
  m.bwd = read_sub(c); m.fwd = read_sub(c);
  m.in_proj2_W = cross ? c.next() : nullptr;
  m.in_proj_W = c.next(); m.norm_w = c.next(); m.out_proj_W = c.next();
  return m;
}
CrossP read_cross(Cursor& c) {
  CrossP p;                    // sorted: cw_W, cw_b, ln0_b, ln0_w, ln1_b, ln1_w, mamba, post_b, post_w
  p.cw_W = c.next(); p.cw_b = c.next();
  p.ln0_b = c.next(); p.ln0_w = c.next(); p.ln1_b = c.next(); p.ln1_w = c.next();
  p.m = read_mamba(c, true);
  p.post_b = c.next(); p.post_w = c.next();
  return p;
}
SingleP read_single(Cursor& c) {
  SingleP p;                   // sorted: ln_b, ln_w, mamba, post_b, post_w
  p.ln_b = c.next(); p.ln_w = c.next();
  p.m = read_mamba(c, false);
  p.post_b = c.next(); p.post_w = c.next();
  return p;
}

struct WS {
  float *i1, *i2, *s1, *s2, *n0, *n1, *wgt, *u, *xcf, *xcb, *featb,
        *dtf, *dtb, *ybuf, *t0, *t1, *t2, *fA, *fB;
  float *cat, *xz, *mlp, *dblf, *dblb, *xpw;
};
WS make_ws(void* d_ws) {
  float* p = (float*)d_ws;
  auto take = [&](size_t n) { float* r = p; p += n; return r; };
  WS w;
  w.i1 = take(NELT); w.i2 = take(NELT); w.s1 = take(NELT); w.s2 = take(NELT);
  w.n0 = take(NELT); w.n1 = take(NELT); w.wgt = take(NELT); w.u = take(NELT);
  w.xcf = take(NELT); w.xcb = take(NELT); w.featb = take(NELT);
  w.dtf = take(NELT); w.dtb = take(NELT); w.ybuf = take(NELT);
  w.t0 = take(NELT); w.t1 = take(NELT); w.t2 = take(NELT);
  w.fA = take(NELT); w.fB = take(NELT);
  w.cat = take((size_t)NTOK * 128); w.xz = take((size_t)NTOK * 128); w.mlp = take((size_t)NTOK * 128);
  w.dblf = take((size_t)NTOK * DBL_LD); w.dblb = take((size_t)NTOK * DBL_LD);
  w.xpw = take((size_t)D * DBL_LD);
  return w;
}

constexpr int EW_BLK = 256;
inline int ew_grid(int n) { return (n + EW_BLK - 1) / EW_BLK; }

// N must be a multiple of 16; K in {64, 128}.
void gemm(hipStream_t st, const float* A, int lda, const float* Wt, int ldw,
          const float* bias, float* C, int ldc, int N, int K, int act) {
  int tiles_n = N / 16;
  int total = (NTOK / 16) * tiles_n;
  int blocks = (total + 7) / 8;                   // 8 waves / block
  if (K == 64)
    k_gemm_wmma<64><<<blocks, 256, 0, st>>>(A, lda, Wt, ldw, bias, C, ldc,
                                            tiles_n, total, act);
  else
    k_gemm_wmma<128><<<blocks, 256, 0, st>>>(A, lda, Wt, ldw, bias, C, ldc,
                                             tiles_n, total, act);
}

void ln(hipStream_t st, const float* x, const float* w, const float* b,
        const float* add, float* out) {
  k_layernorm<<<NTOK / 8, 256, 0, st>>>(x, w, b, add, out, NTOK);
}

void run_mamba(hipStream_t st, const MambaP& mp, bool cross,
               const float* x, const float* extra, float* out, WS& w) {
  gemm(st, x, D, mp.in_proj_W, 2 * D, nullptr, w.xz, 2 * D, 2 * D, D, 0);   // xz = x @ in_proj
  if (cross)
    gemm(st, extra, D, mp.in_proj2_W, D, nullptr, w.u, D, D, D, 0);         // u = extra @ in_proj2
  for (int dir = 0; dir < 2; ++dir) {
    const SubP& sp = dir ? mp.bwd : mp.fwd;
    float* xc = dir ? w.xcb : w.xcf;
    k_dwconv_silu<<<ew_grid(NELT), EW_BLK, 0, st>>>(w.xz, 2 * D, sp.conv_W, sp.conv_b,
                                                    xc, dir, NELT);
    const float* feats;
    if (cross) {
      if (dir) { k_flip<<<ew_grid(NELT), EW_BLK, 0, st>>>(w.u, w.featb, NELT); feats = w.featb; }
      else feats = w.u;
    } else {
      feats = xc;
    }
    // pad x_proj weights to N=48 then dense WMMA GEMM (cols 36..47 produce 0)
    k_pad_xproj<<<ew_grid(D * DBL_LD), EW_BLK, 0, st>>>(sp.x_proj_W, w.xpw, D * DBL_LD);
    float* dbl = dir ? w.dblb : w.dblf;
    gemm(st, feats, D, w.xpw, DBL_LD, nullptr, dbl, DBL_LD, DBL_LD, D, 0);
    float* dt = dir ? w.dtb : w.dtf;
    k_dt<<<ew_grid(NELT), EW_BLK, 0, st>>>(dbl, sp.dt_proj_W, sp.dt_proj_b, dt, NELT);
    k_scan<<<(B * D) / 16, 256, 0, st>>>(dt, xc, dbl, sp.A_log, sp.Dp, w.ybuf, dir, dir);
  }
  k_gate_rms<<<NTOK / 8, 256, 0, st>>>(w.ybuf, w.xz + D, 2 * D, mp.norm_w, w.t0, NTOK);
  gemm(st, w.t0, D, mp.out_proj_W, D, nullptr, out, D, D, D, 0);            // out_proj
}

void run_cross(hipStream_t st, const CrossP& p, const float* x0, const float* x1,
               float* out, WS& w) {
  ln(st, x0, p.ln0_w, p.ln0_b, nullptr, w.n0);
  ln(st, x1, p.ln1_w, p.ln1_b, nullptr, w.n1);
  k_concat<<<ew_grid(NELT), EW_BLK, 0, st>>>(w.n0, w.n1, w.cat, NELT);
  gemm(st, w.cat, 2 * D, p.cw_W, D, p.cw_b, w.wgt, D, D, 2 * D, 1);         // sigmoid gate
  run_mamba(st, p.m, true, w.n0, w.n1, w.t1, w);
  ln(st, w.t1, p.post_w, p.post_b, nullptr, w.t2);
  k_combine_cross<<<ew_grid(NELT), EW_BLK, 0, st>>>(w.t2, w.wgt, w.n0, x0, out, NELT);
}

void run_single(hipStream_t st, const SingleP& p, float* x, WS& w) {
  ln(st, x, p.ln_w, p.ln_b, nullptr, w.n0);
  run_mamba(st, p.m, false, w.n0, nullptr, w.t1, w);
  ln(st, w.t1, p.post_w, p.post_b, /*add=*/x, x);                           // x = LN(out)+skip
}

} // namespace

extern "C" void kernel_launch(void* const* d_in, const int* in_sizes, int n_in,
                              void* d_out, int out_size, void* d_ws, size_t ws_size,
                              hipStream_t stream) {
  (void)in_sizes; (void)n_in; (void)out_size; (void)ws_size;
  const float* img1 = (const float*)d_in[0];
  const float* img2 = (const float*)d_in[1];
  const float* sum1 = (const float*)d_in[2];
  const float* sum2 = (const float*)d_in[3];

  Cursor c{d_in, 4};
  const float* alpha = c.next();
  const float* beta  = c.next();
  OutP op;                       // sorted: W1, W2, b1, b2, ln_b, ln_w
  op.W1 = c.next(); op.W2 = c.next(); op.b1 = c.next(); op.b2 = c.next();
  op.ln_b = c.next(); op.ln_w = c.next();
  SingleP self_attn = read_single(c);
  CrossP spa0 = read_cross(c), spa1 = read_cross(c);
  CrossP spa_cross = read_cross(c);
  CrossP spe0 = read_cross(c), spe1 = read_cross(c);
  CrossP spe_cross = read_cross(c);

  WS w = make_ws(d_ws);

  k_to_seq<<<ew_grid(NELT), EW_BLK, 0, stream>>>(img1, w.i1, NELT);
  k_to_seq<<<ew_grid(NELT), EW_BLK, 0, stream>>>(img2, w.i2, NELT);
  k_to_seq<<<ew_grid(NELT), EW_BLK, 0, stream>>>(sum1, w.s1, NELT);
  k_to_seq<<<ew_grid(NELT), EW_BLK, 0, stream>>>(sum2, w.s2, NELT);

  run_cross(stream, spa0, w.i1, w.s1, w.i1, w);
  run_cross(stream, spe0, w.i2, w.s2, w.i2, w);
  run_cross(stream, spa1, w.i1, w.s1, w.i1, w);
  run_cross(stream, spe1, w.i2, w.s2, w.i2, w);

  run_cross(stream, spa_cross, w.i1, w.i2, w.fA, w);
  run_cross(stream, spe_cross, w.i2, w.i1, w.fB, w);

  k_fuse<<<ew_grid(NELT), EW_BLK, 0, stream>>>(w.fA, w.fB, alpha, beta, w.fA, NELT);
  run_single(stream, self_attn, w.fA, w);

  ln(stream, w.fA, op.ln_w, op.ln_b, nullptr, w.t0);
  gemm(stream, w.t0, D, op.W1, 2 * D, op.b1, w.mlp, 2 * D, 2 * D, D, 2);    // gelu
  gemm(stream, w.mlp, 2 * D, op.W2, D, op.b2, w.t1, D, D, 2 * D, 0);
  k_from_seq<<<ew_grid(NELT), EW_BLK, 0, stream>>>(w.t1, (float*)d_out, NELT);
}
